// Attention_58067957842238
// MI455X (gfx1250) — matrix operand order repaired and checked by reading, not compiled, not run on hardware
//
#include <hip/hip_runtime.h>

// ---------------------------------------------------------------------------
// CDNA5 (gfx1250) fused attention, round 2: vectorized fragment traffic.
//   - A fragments: 2x ds_load_b128 from [m][k] tiles (contiguous 8-elem runs)
//   - B fragments: packed-fragment LDS layout -> one aligned 32B read/lane
//   - global staging: bf16x8 / float4 vector loads
// wave32, 128 threads (4 waves) per WG, 64x64 tiles,
// v_wmma_f32_16x16x32_bf16 with f32 accumulation.
// ---------------------------------------------------------------------------

typedef __bf16 bf16;
typedef __bf16 v16bf  __attribute__((ext_vector_type(16)));
typedef __bf16 bf16x8 __attribute__((ext_vector_type(8)));
typedef __bf16 bf16x4 __attribute__((ext_vector_type(4)));
typedef float  v8f    __attribute__((ext_vector_type(8)));

#define BATCH  2
#define HEADS  4
#define DHEAD  64
#define DIM    256
#define SEQ    4096          // 64 * 64 spatial positions
#define QKV_CH 768

__device__ __forceinline__ bf16 f2bf(float f) { return (bf16)f; }

// A-matrix fragment (16x32 bf16, MxK) from [m][k] LDS tile, ld elements/row.
// lane l: M=l&15, kh=(l>>4)*16; elems 0..15 -> K=k0+kh+j.
// One contiguous 32B run -> 2x ds_load_b128.
__device__ __forceinline__ v16bf lds_load_A(const bf16* base, int row, int k0, int ld) {
  const int lane = threadIdx.x & 31;
  const int m    = lane & 15;
  const int kh   = (lane >> 4) << 4;
  const bf16* p  = base + (row + m) * ld + k0 + kh;
  bf16x8 lo = *(const bf16x8*)p;
  bf16x8 hi = *(const bf16x8*)(p + 8);
  return __builtin_shufflevector(lo, hi, 0, 1, 2, 3, 4, 5, 6, 7,
                                 8, 9, 10, 11, 12, 13, 14, 15);
}

// Packed B layout: for sub-tile (k_blk: 32 K-values, n_blk: 16 N-values),
// lane l's 16 elements live contiguously at (n_blk*KB + k_blk)*512 + l*16.
// Element j of lane l is (K = k_blk*32 + (l>>4)*16 + j, N = n_blk*16 + (l&15)).
__device__ __forceinline__ int bpack_idx(int k, int n, int KB) {
  return ((n >> 4) * KB + (k >> 5)) * 512 + ((k >> 4) & 1) * 256 +
         (n & 15) * 16 + (k & 15);
}

__device__ __forceinline__ v16bf lds_load_Bp(const bf16* base, int k_blk, int n_blk,
                                             int KB) {
  const int lane = threadIdx.x & 31;
  return *(const v16bf*)(base + (n_blk * KB + k_blk) * 512 + lane * 16);
}

__device__ __forceinline__ v8f wmma_bf16(v16bf a, v16bf b, v8f c) {
  return __builtin_amdgcn_wmma_f32_16x16x32_bf16(false, a, false, b, (short)0, c,
                                                 false, false);
}

// ---------------------------------------------------------------------------
// Generic 64x64-tile GEMM:  Out[b] = A(MxK,f32) * B[b](KxN)  (+ bias)
// ---------------------------------------------------------------------------
template <bool B_IS_BF16, bool OUT_F32_BIAS>
__global__ __launch_bounds__(128)
void gemm_kernel(const float* __restrict__ A, const void* __restrict__ Bsrc,
                 const float* __restrict__ bias, void* __restrict__ Out,
                 int M, int N, int K, long strideB, long strideO) {
  const int n0   = blockIdx.x * 64;
  const int m0   = blockIdx.y * 64;
  const int b    = blockIdx.z;
  const int tid  = threadIdx.x;
  const int wave = tid >> 5;
  const int lane = tid & 31;

  __shared__ __align__(32) bf16 sA[64 * 32];   // [m][k]
  __shared__ __align__(32) bf16 sB[32 * 64];   // packed-fragment, KB=1

  const float* Bf = (const float*)Bsrc + (long)b * strideB;
  const bf16*  Bh = (const bf16*)Bsrc + (long)b * strideB;

  v8f acc[4] = {};

  for (int k0 = 0; k0 < K; k0 += 32) {
    // stage A tile 64x32 (f32 -> bf16): float4 loads, bf16x4 stores
#pragma unroll
    for (int i = 0; i < 4; ++i) {
      int idx = (tid + i * 128) * 4;        // 0..2044, step 4
      int m = idx >> 5, k = idx & 31;
      float4 v = *(const float4*)&A[(long)(m0 + m) * K + k0 + k];
      bf16x4 h = {f2bf(v.x), f2bf(v.y), f2bf(v.z), f2bf(v.w)};
      *(bf16x4*)&sA[m * 32 + k] = h;
    }
    // stage B tile 32x64 into packed layout
#pragma unroll
    for (int i = 0; i < 4; ++i) {
      int idx = (tid + i * 128) * 4;
      int k = idx >> 6, n = idx & 63;       // 4 consecutive n
      float e[4];
      if constexpr (B_IS_BF16) {
        bf16x4 v = *(const bf16x4*)&Bh[(long)(k0 + k) * N + n0 + n];
        e[0] = (float)v[0]; e[1] = (float)v[1]; e[2] = (float)v[2]; e[3] = (float)v[3];
      } else {
        float4 v = *(const float4*)&Bf[(long)(k0 + k) * N + n0 + n];
        e[0] = v.x; e[1] = v.y; e[2] = v.z; e[3] = v.w;
      }
#pragma unroll
      for (int j = 0; j < 4; ++j)
        sB[bpack_idx(k, n + j, 1)] = f2bf(e[j]);
    }
    __syncthreads();

    v16bf a = lds_load_A(sA, 16 * wave, 0, 32);
#pragma unroll
    for (int n = 0; n < 4; ++n)
      acc[n] = wmma_bf16(a, lds_load_Bp(sB, 0, n, 1), acc[n]);
    __syncthreads();
  }

  // epilogue: D layout -> row = r + 8*(lane>>4), col = lane&15
  const int colL = lane & 15;
  const int rhi  = (lane >> 4) * 8;
#pragma unroll
  for (int n = 0; n < 4; ++n) {
#pragma unroll
    for (int r = 0; r < 8; ++r) {
      int m = m0 + 16 * wave + rhi + r;
      int c = n0 + n * 16 + colL;
      float v = acc[n][r];
      if constexpr (OUT_F32_BIAS)
        ((float*)Out)[(long)b * strideO + (long)m * N + c] = v + bias[m];
      else
        ((bf16*)Out)[(long)b * strideO + (long)m * N + c] = f2bf(v);
    }
  }
}

// ---------------------------------------------------------------------------
// Fused flash attention. qkv: (b, 768, 4096) bf16, q/k/v at channel offsets
// 0/256/512 (+ h*64). One workgroup = one (b, h, 64-query block).
// ---------------------------------------------------------------------------
__global__ __launch_bounds__(128)
void attn_kernel(const bf16* __restrict__ qkv, bf16* __restrict__ o) {
  const int x0   = blockIdx.x * 64;
  const int h    = blockIdx.y;
  const int b    = blockIdx.z;
  const int tid  = threadIdx.x;
  const int wave = tid >> 5;
  const int lane = tid & 31;

  const long baseQ = ((long)b * QKV_CH +   0 + h * DHEAD) * SEQ;
  const long baseK = ((long)b * QKV_CH + 256 + h * DHEAD) * SEQ;
  const long baseV = ((long)b * QKV_CH + 512 + h * DHEAD) * SEQ;

  __shared__ __align__(32) bf16  sQ[64 * 64];   // [x][c]  (Q^T, pre-scaled)
  __shared__ __align__(32) bf16  sK[64 * 64];   // packed B: K=c (KB=2), N=y
  __shared__ __align__(32) bf16  sV[64 * 64];   // packed B: K=y (KB=2), N=c
  __shared__ __align__(16) float sS[64 * 64];   // scores
  __shared__ __align__(32) bf16  sP[64 * 64];   // [x][y] exp(scores - m)
  __shared__ float sM[64], sL[64], sC[64];

  const float scale = 0.125f;                    // DIM_HEAD^-0.5

  // stage Q transposed + scaled: bf16x8 global loads along x, b16 scatter
#pragma unroll
  for (int i = 0; i < 4; ++i) {
    int ci = tid + i * 128;                      // 0..511
    int c = ci >> 3, x8 = (ci & 7) * 8;
    bf16x8 v = *(const bf16x8*)&qkv[baseQ + (long)c * SEQ + x0 + x8];
#pragma unroll
    for (int j = 0; j < 8; ++j)
      sQ[(x8 + j) * 64 + c] = f2bf((float)v[j] * scale);
  }
  if (tid < 64) { sM[tid] = -1e30f; sL[tid] = 0.f; }
  __syncthreads();

  const v16bf aQ0 = lds_load_A(sQ, 16 * wave, 0, 64);
  const v16bf aQ1 = lds_load_A(sQ, 16 * wave, 32, 64);

  v8f oacc[4] = {};
  const int colL = lane & 15;
  const int rhi  = (lane >> 4) * 8;

  for (int kv0 = 0; kv0 < SEQ; kv0 += 64) {
    // stage K (packed, K=c, N=y) and V (packed, K=y, N=c)
#pragma unroll
    for (int i = 0; i < 4; ++i) {
      int ci = tid + i * 128;
      int c = ci >> 3, y8 = (ci & 7) * 8;
      bf16x8 kv = *(const bf16x8*)&qkv[baseK + (long)c * SEQ + kv0 + y8];
      bf16x8 vv = *(const bf16x8*)&qkv[baseV + (long)c * SEQ + kv0 + y8];
#pragma unroll
      for (int j = 0; j < 8; ++j)
        sK[bpack_idx(c, y8 + j, 2)] = kv[j];
      // 8 consecutive y = 8 consecutive packed elements -> one 16B store
      *(bf16x8*)&sV[bpack_idx(y8, c, 2)] = vv;
    }
    __syncthreads();

    // S = (Q*scale) K : per wave a 16x64 strip, K-dim 64 as two k=32 steps
#pragma unroll
    for (int n = 0; n < 4; ++n) {
      v8f s = {};
      s = wmma_bf16(aQ0, lds_load_Bp(sK, 0, n, 2), s);
      s = wmma_bf16(aQ1, lds_load_Bp(sK, 1, n, 2), s);
#pragma unroll
      for (int r = 0; r < 8; ++r)
        sS[(16 * wave + rhi + r) * 64 + n * 16 + colL] = s[r];
    }
    __syncthreads();

    // online softmax: one thread per query row, float4 scans
    if (tid < 64) {
      const float4* row4 = (const float4*)&sS[tid * 64];
      float mNew = sM[tid];
#pragma unroll
      for (int j = 0; j < 16; ++j) {
        float4 v = row4[j];
        mNew = fmaxf(mNew, fmaxf(fmaxf(v.x, v.y), fmaxf(v.z, v.w)));
      }
      float corr = __expf(sM[tid] - mNew);
      float l = sL[tid] * corr;
#pragma unroll
      for (int j = 0; j < 16; ++j) {
        float4 v = row4[j];
        float p0 = __expf(v.x - mNew), p1 = __expf(v.y - mNew);
        float p2 = __expf(v.z - mNew), p3 = __expf(v.w - mNew);
        l += (p0 + p1) + (p2 + p3);
        bf16x4 p = {f2bf(p0), f2bf(p1), f2bf(p2), f2bf(p3)};
        *(bf16x4*)&sP[tid * 64 + j * 4] = p;
      }
      sM[tid] = mNew; sL[tid] = l; sC[tid] = corr;
    }
    __syncthreads();

    // rescale running O accumulators by exp(m_old - m_new) per row
#pragma unroll
    for (int r = 0; r < 8; ++r) {
      float f = sC[16 * wave + rhi + r];
#pragma unroll
      for (int n = 0; n < 4; ++n) oacc[n][r] *= f;
    }

    // O += P V^T : A = sP (16x64 per wave), B = sV packed
    const v16bf aP0 = lds_load_A(sP, 16 * wave, 0, 64);
    const v16bf aP1 = lds_load_A(sP, 16 * wave, 32, 64);
#pragma unroll
    for (int n = 0; n < 4; ++n) {
      oacc[n] = wmma_bf16(aP0, lds_load_Bp(sV, 0, n, 2), oacc[n]);
      oacc[n] = wmma_bf16(aP1, lds_load_Bp(sV, 1, n, 2), oacc[n]);
    }
    __syncthreads();
  }

  // finalize: divide by l, write o[b][h*64+c][x] (bf16, feeds out-projection)
#pragma unroll
  for (int r = 0; r < 8; ++r) {
    int row = 16 * wave + rhi + r;
    float invl = 1.0f / sL[row];
    int x = x0 + row;
#pragma unroll
    for (int n = 0; n < 4; ++n) {
      int c = n * 16 + colL;
      o[((long)b * DIM + h * DHEAD + c) * SEQ + x] = f2bf(oacc[n][r] * invl);
    }
  }
}

// ---------------------------------------------------------------------------
extern "C" void kernel_launch(void* const* d_in, const int* in_sizes, int n_in,
                              void* d_out, int out_size, void* d_ws, size_t ws_size,
                              hipStream_t stream) {
  (void)in_sizes; (void)n_in; (void)out_size; (void)ws_size;
  const float* x     = (const float*)d_in[0];   // (2, 256, 64, 64)
  const float* w_qkv = (const float*)d_in[1];   // (768, 256)
  const float* w_out = (const float*)d_in[2];   // (256, 256)
  const float* b_out = (const float*)d_in[3];   // (256,)
  float* out = (float*)d_out;                   // (2, 256, 64, 64) f32

  // workspace: qkv bf16 (2*768*4096) | attn-out bf16 (2*256*4096)
  bf16* qkv  = (bf16*)d_ws;
  bf16* attn = (bf16*)((char*)d_ws + (size_t)BATCH * QKV_CH * SEQ * sizeof(bf16));

  dim3 blk(128);

  // 1) QKV projection: qkv[b] = w_qkv (768x256) * x[b] (256x4096), bf16 out
  gemm_kernel<false, false><<<dim3(SEQ / 64, QKV_CH / 64, BATCH), blk, 0, stream>>>(
      w_qkv, x, nullptr, qkv, QKV_CH, SEQ, DIM,
      (long)DIM * SEQ, (long)QKV_CH * SEQ);

  // 2) fused flash attention per (b, h, 64-query block)
  attn_kernel<<<dim3(SEQ / 64, HEADS, BATCH), blk, 0, stream>>>(qkv, attn);

  // 3) output projection + bias: out[b] = w_out (256x256) * attn[b] + b_out
  gemm_kernel<true, true><<<dim3(SEQ / 64, DIM / 64, BATCH), blk, 0, stream>>>(
      w_out, attn, b_out, out, DIM, SEQ, DIM,
      (long)DIM * SEQ, (long)DIM * SEQ);
}
